// SolidsHead_68384469286984
// MI455X (gfx1250) — compile-verified, matrix operand order checked by reading, not run on hardware
//
#include <hip/hip_runtime.h>
#include <hip/hip_bf16.h>
#include <math.h>

// ---------------- problem constants ----------------
#define P_PIX   32768      // 2*128*128
#define EMBED   256
#define HIDDEN  512
#define KTOP    5
#define NLIB    16384
#define NBANDS  224
#define FUSEDK  288        // 261 padded to multiple of 32
// output layout (floats): dominant | abundances | recon | cand_sims
#define OFF_DOM   0
#define OFF_AB    (P_PIX)
#define OFF_REC   (OFF_AB + P_PIX*7)
#define OFF_SIMS  (OFF_REC + P_PIX*NBANDS)

typedef _Float16 h16;
typedef __attribute__((ext_vector_type(8)))  _Float16 v8h;
typedef __attribute__((ext_vector_type(16))) _Float16 v16h;
typedef __attribute__((ext_vector_type(8)))  float    v8f;
typedef __attribute__((ext_vector_type(4)))  float    v4f;

// ------------------------------------------------------------------
// Fragment loader for v_wmma_f32_16x16x32_f16.
// A is M(rows striped over lanes) x K; B is supplied as Bt[N][K] so the
// same loader works for both (N plays the role of M).
// Per ISA 7.12.2 (16-bit A 16x32): lanes 0-15 hold row r=lane, K chunks
// {0..7} and {16..23}; lanes 16-31 hold row r=lane-16, K chunks {8..15}
// and {24..31}.  Two 16-byte LDS loads per lane.
// ------------------------------------------------------------------
__device__ __forceinline__ v16h load_frag16(const h16* base, int ld) {
  int lane = threadIdx.x & 31;
  int r  = lane & 15;
  int hi = lane >> 4;
  const h16* p = base + r * ld + hi * 8;
  v8h a = *(const v8h*)p;
  v8h b = *(const v8h*)(p + 16);
  v16h f;
#pragma unroll
  for (int i = 0; i < 8; ++i) { f[i] = a[i]; f[i + 8] = b[i]; }
  return f;
}

__device__ __forceinline__ v8f wmma_f16(v16h a, v16h b, v8f c) {
  return __builtin_amdgcn_wmma_f32_16x16x32_f16(false, a, false, b,
                                                (short)0, c, false, false);
}

// ---------------- weight convert + transpose: out[n*Kout+k] = in[k*N+n] ----
__global__ __launch_bounds__(256) void wtrans_kernel(
    const float* __restrict__ in, h16* __restrict__ out,
    int Kin, int Kout, int N) {
  int i = blockIdx.x * 256 + threadIdx.x;
  if (i >= N * Kout) return;
  int n = i / Kout, k = i - n * Kout;
  out[i] = (k < Kin) ? (h16)in[k * N + n] : (h16)0.0f;
}

// ---------------- normalize library rows -> f16 ----------------
__global__ __launch_bounds__(256) void prep_lib_kernel(
    const float* __restrict__ lib, h16* __restrict__ lib16) {
  int wave = threadIdx.x >> 5, lane = threadIdx.x & 31;
  int row = blockIdx.x * 8 + wave;
  if (row >= NLIB) return;
  const float* src = lib + row * EMBED + lane * 8;
  v4f x0 = *(const v4f*)src;
  v4f x1 = *(const v4f*)(src + 4);
  float ss = 0.f;
#pragma unroll
  for (int i = 0; i < 4; ++i) ss += x0[i] * x0[i] + x1[i] * x1[i];
#pragma unroll
  for (int off = 16; off > 0; off >>= 1) ss += __shfl_xor(ss, off, 32);
  float sc = 1.0f / (sqrtf(ss) + 1e-12f);
  v8h o;
#pragma unroll
  for (int i = 0; i < 4; ++i) { o[i] = (h16)(x0[i] * sc); o[i + 4] = (h16)(x1[i] * sc); }
  *(v8h*)(lib16 + row * EMBED + lane * 8) = o;
}

// ------- normalize feature rows -> fn16; raw f16 -> fused[:,0:256]; pad 0 ---
__global__ __launch_bounds__(256) void prep_feat_kernel(
    const float* __restrict__ feat, h16* __restrict__ fn16,
    h16* __restrict__ fused) {
  int wave = threadIdx.x >> 5, lane = threadIdx.x & 31;
  int row = blockIdx.x * 8 + wave;
  if (row >= P_PIX) return;
  const float* src = feat + row * EMBED + lane * 8;
  v4f x0 = *(const v4f*)src;
  v4f x1 = *(const v4f*)(src + 4);
  float ss = 0.f;
#pragma unroll
  for (int i = 0; i < 4; ++i) ss += x0[i] * x0[i] + x1[i] * x1[i];
#pragma unroll
  for (int off = 16; off > 0; off >>= 1) ss += __shfl_xor(ss, off, 32);
  float sc = 1.0f / (sqrtf(ss) + 1e-12f);
  v8h on, or_;
#pragma unroll
  for (int i = 0; i < 4; ++i) {
    on[i]     = (h16)(x0[i] * sc); on[i + 4]  = (h16)(x1[i] * sc);
    or_[i]    = (h16)x0[i];        or_[i + 4] = (h16)x1[i];
  }
  *(v8h*)(fn16  + row * EMBED  + lane * 8) = on;
  *(v8h*)(fused + row * FUSEDK + lane * 8) = or_;
  if (lane < 4) {  // zero pad cols 256..287 (scores written later)
    v8h z = {};
    *(v8h*)(fused + row * FUSEDK + EMBED + lane * 8) = z;
  }
}

// ---------------- top-k insert helper ----------------
__device__ __forceinline__ void topk_insert(float (&tv)[KTOP], int (&ti)[KTOP],
                                            float v, int id) {
  if (v <= tv[KTOP - 1]) return;
  tv[KTOP - 1] = v; ti[KTOP - 1] = id;
#pragma unroll
  for (int s = KTOP - 1; s > 0; --s) {
    if (tv[s] > tv[s - 1]) {
      float tf = tv[s]; tv[s] = tv[s - 1]; tv[s - 1] = tf;
      int t2 = ti[s]; ti[s] = ti[s - 1]; ti[s - 1] = t2;
    }
  }
}

// ------------------------------------------------------------------
// Fused cosine-similarity GEMM + top-5.
// Block = 256 threads (8 waves).  Output tile: 64 pixel rows x 64 lib cols
// per chunk; 256 chunks cover NLIB.
//  - A fragments (16x256 strip per wave) hoisted into registers once.
//  - B staged global->reg->LDS with a ping-pong register double buffer so
//    the L2 latency of chunk i+1 overlaps the 16 WMMAs of chunk i.
//  - B fragments software-pipelined one k-step ahead in distinct registers
//    so >=4 ds_load_b128 stay in flight behind each WMMA pair.
//  - Per-chunk 64x64 f32 C staged in LDS, folded into per-thread top-5.
// ------------------------------------------------------------------
#define SIMS_LDS_B   (64 * EMBED * 2)        // 32 KB (also A-stage at start)
#define SIMS_LDS_C   (64 * 64 * 4)           // 16 KB
__global__ __launch_bounds__(256) void sims_topk_kernel(
    const h16* __restrict__ fn16, const h16* __restrict__ lib16,
    float* __restrict__ out_sims, int* __restrict__ cand_ids) {
  __shared__ __align__(16) unsigned char smem[SIMS_LDS_B + SIMS_LDS_C];
  h16*   Bs = (h16*)smem;
  float* Cs = (float*)(smem + SIMS_LDS_B);
  // merge scratch aliases the B region (only used after the main loop)
  float* mv = (float*)smem;
  int*   mi = (int*)(smem + 64 * 4 * KTOP * 4);

  const int tid  = threadIdx.x;
  const int wave = tid >> 5, lane = tid & 31;
  const int pbase = blockIdx.x * 64;
  const int rt = wave & 3;   // row tile 0..3 (16 rows each)
  const int cw = wave >> 2;  // col half 0..1 (2 x 16-col tiles each)
  const int srow = tid >> 5;        // staging base row 0..7 (stride 8)
  const int scol = (tid & 31) * 8;  // staging col in halves

  // ---- prologue: issue chunk-0 B loads into bufA ----
  v8h bufA[8], bufB[8];
#pragma unroll
  for (int i = 0; i < 8; ++i)
    bufA[i] = *(const v8h*)(lib16 + (srow + 8 * i) * EMBED + scol);

  // ---- stage A tile (reg pipelined), hoist fragments, free the region ----
  {
    v8h areg[8];
#pragma unroll
    for (int i = 0; i < 8; ++i)
      areg[i] = *(const v8h*)(fn16 + (pbase + srow + 8 * i) * EMBED + scol);
#pragma unroll
    for (int i = 0; i < 8; ++i)
      *(v8h*)((h16*)smem + (srow + 8 * i) * EMBED + scol) = areg[i];
  }
  __syncthreads();
  v16h afrag[8];
#pragma unroll
  for (int kk = 0; kk < 8; ++kk)
    afrag[kk] = load_frag16((h16*)smem + (rt * 16) * EMBED + kk * 32, EMBED);
  __syncthreads();

  float topv[KTOP]; int topi[KTOP];
#pragma unroll
  for (int j = 0; j < KTOP; ++j) { topv[j] = -1e30f; topi[j] = 0; }

  const int myrow = tid & 63;
  const int mygrp = tid >> 6;  // 4 col groups of 16

  const h16* b0base = Bs + (cw * 32) * EMBED;        // col tile ct0
  const h16* b1base = Bs + (cw * 32 + 16) * EMBED;   // col tile ct1

  auto chunk_step = [&](int nb, v8h (&use)[8], v8h (&pre)[8]) {
    // commit this chunk's staged registers to LDS
#pragma unroll
    for (int i = 0; i < 8; ++i)
      *(v8h*)(Bs + (srow + 8 * i) * EMBED + scol) = use[i];
    __syncthreads();
    // issue next chunk's global loads (wait lands at next ds_store batch)
    int nn = nb + 64;
    if (nn < NLIB) {
#pragma unroll
      for (int i = 0; i < 8; ++i)
        pre[i] = *(const v8h*)(lib16 + (nn + srow + 8 * i) * EMBED + scol);
    }
    // 16 WMMAs, B fragments pipelined one k-step ahead
    v8f acc0 = {}, acc1 = {};
    v16h b0 = load_frag16(b0base, EMBED);
    v16h b1 = load_frag16(b1base, EMBED);
#pragma unroll
    for (int kk = 0; kk < 8; ++kk) {
      v16h cb0 = b0, cb1 = b1;
      if (kk < 7) {
        b0 = load_frag16(b0base + (kk + 1) * 32, EMBED);
        b1 = load_frag16(b1base + (kk + 1) * 32, EMBED);
      }
      acc0 = wmma_f16(afrag[kk], cb0, acc0);
      acc1 = wmma_f16(afrag[kk], cb1, acc1);
    }
    // C layout: lanes 0-15 -> M=v, lanes 16-31 -> M=v+8; N = lane&15
    int hi = lane >> 4;
    int r0 = rt * 16 + hi * 8;
    int n0 = cw * 32 + (lane & 15);
#pragma unroll
    for (int v = 0; v < 8; ++v) {
      Cs[(r0 + v) * 64 + n0]      = acc0[v];
      Cs[(r0 + v) * 64 + n0 + 16] = acc1[v];
    }
    __syncthreads();
#pragma unroll
    for (int j = 0; j < 16; ++j) {
      int c = mygrp * 16 + j;
      topk_insert(topv, topi, Cs[myrow * 64 + c], nb + c);
    }
  };

  for (int nb = 0; nb < NLIB; nb += 128) {
    chunk_step(nb,      bufA, bufB);
    chunk_step(nb + 64, bufB, bufA);
  }

  // ---- merge 4 partial top-5 lists per row ----
#pragma unroll
  for (int j = 0; j < KTOP; ++j) {
    mv[(myrow * 4 + mygrp) * KTOP + j] = topv[j];
    mi[(myrow * 4 + mygrp) * KTOP + j] = topi[j];
  }
  __syncthreads();
  if (tid < 64) {
    float vals[4 * KTOP]; int ids[4 * KTOP];
#pragma unroll
    for (int g = 0; g < 4; ++g)
#pragma unroll
      for (int j = 0; j < KTOP; ++j) {
        vals[g * KTOP + j] = mv[(tid * 4 + g) * KTOP + j];
        ids[g * KTOP + j]  = mi[(tid * 4 + g) * KTOP + j];
      }
    int p = pbase + tid;
    for (int s = 0; s < KTOP; ++s) {
      int bi = 0;
      for (int t = 1; t < 4 * KTOP; ++t)
        if (vals[t] > vals[bi]) bi = t;
      out_sims[p * KTOP + s] = vals[bi];
      cand_ids[p * KTOP + s] = ids[bi];
      vals[bi] = -1e30f;
    }
  }
}

// ------------------------------------------------------------------
// WMMA hidden layer: H = relu(A[P,K] @ Bt^T + bias), H in f16 [P,512].
// Bt stored [512][K] f16.  Tile 64x32, K in {256, 288}.
// ------------------------------------------------------------------
#define HID_LDS ((64 + 32) * FUSEDK * 2)
__global__ __launch_bounds__(256) void mlp_hidden_kernel(
    const h16* __restrict__ A, int lda, const h16* __restrict__ Bt, int K,
    const float* __restrict__ bias, h16* __restrict__ H) {
  __shared__ __align__(16) unsigned char smem[HID_LDS];
  h16* As = (h16*)smem;
  h16* Bs = As + 64 * FUSEDK;  // fixed budget, runtime ld = K

  int tid = threadIdx.x;
  int wave = tid >> 5, lane = tid & 31;
  int pbase = blockIdx.x * 64;
  int nbase = blockIdx.y * 32;
  int kv = K / 8;

  for (int idx = tid; idx < 64 * kv; idx += 256) {
    int r = idx / kv, c = idx - r * kv;
    *(v8h*)(As + r * K + c * 8) = *(const v8h*)(A + (pbase + r) * lda + c * 8);
  }
  for (int idx = tid; idx < 32 * kv; idx += 256) {
    int r = idx / kv, c = idx - r * kv;
    *(v8h*)(Bs + r * K + c * 8) = *(const v8h*)(Bt + (nbase + r) * K + c * 8);
  }
  __syncthreads();

  int rt = wave & 3, ct = wave >> 2;
  v8f acc = {};
  for (int kk = 0; kk < K / 32; ++kk) {
    v16h a = load_frag16(As + (rt * 16) * K + kk * 32, K);
    v16h b = load_frag16(Bs + (ct * 16) * K + kk * 32, K);
    acc = wmma_f16(a, b, acc);
  }
  int hi = lane >> 4;
  int n = nbase + ct * 16 + (lane & 15);
  float bv = bias[n];
#pragma unroll
  for (int v = 0; v < 8; ++v) {
    int row = pbase + rt * 16 + v + 8 * hi;
    H[row * HIDDEN + n] = (h16)fmaxf(acc[v] + bv, 0.0f);
  }
}

// -------- scores = H @ w2_s + b2_s, written into fused[:,256:261] ---------
__global__ __launch_bounds__(256) void score_out_kernel(
    const h16* __restrict__ H, const float* __restrict__ w2,
    const float* __restrict__ b2, h16* __restrict__ fused) {
  int i = blockIdx.x * 256 + threadIdx.x;
  if (i >= P_PIX * KTOP) return;
  int p = i / KTOP, o = i - p * KTOP;
  const h16* h = H + p * HIDDEN;
  float s = b2[o];
  for (int j8 = 0; j8 < HIDDEN; j8 += 8) {
    v8h hv = *(const v8h*)(h + j8);
#pragma unroll
    for (int t = 0; t < 8; ++t) s += (float)hv[t] * w2[(j8 + t) * KTOP + o];
  }
  fused[p * FUSEDK + EMBED + o] = (h16)s;
}

// -------- final: logits -> softmax -> abund / dominant / recon ----------
__global__ __launch_bounds__(256) void final_kernel(
    const h16* __restrict__ H, const float* __restrict__ w2a,
    const float* __restrict__ b2a, const int* __restrict__ cand_ids,
    const float* __restrict__ protos, float* __restrict__ out) {
  int wave = threadIdx.x >> 5, lane = threadIdx.x & 31;
  int p = blockIdx.x * 8 + wave;
  if (p >= P_PIX) return;
  const h16* h = H + p * HIDDEN;
  float l[7];
#pragma unroll
  for (int o = 0; o < 7; ++o) l[o] = 0.f;
  for (int j = lane; j < HIDDEN; j += 32) {
    float hv = (float)h[j];
    const float* wr = w2a + j * 7;
#pragma unroll
    for (int o = 0; o < 7; ++o) l[o] += hv * wr[o];
  }
#pragma unroll
  for (int o = 0; o < 7; ++o) {
#pragma unroll
    for (int off = 16; off > 0; off >>= 1) l[o] += __shfl_xor(l[o], off, 32);
    l[o] += b2a[o];
  }
  // softmax (all lanes redundantly)
  float m = l[0];
#pragma unroll
  for (int o = 1; o < 7; ++o) m = fmaxf(m, l[o]);
  float a[7], sum = 0.f;
#pragma unroll
  for (int o = 0; o < 7; ++o) { a[o] = expf(l[o] - m); sum += a[o]; }
  float inv = 1.0f / sum;
#pragma unroll
  for (int o = 0; o < 7; ++o) a[o] *= inv;

  int ids[KTOP];
#pragma unroll
  for (int j = 0; j < KTOP; ++j) ids[j] = cand_ids[p * KTOP + j];
  int best = 0;
#pragma unroll
  for (int j = 1; j < KTOP; ++j) if (a[j] > a[best]) best = j;

  if (lane == 0) {
#pragma unroll
    for (int o = 0; o < 7; ++o) out[OFF_AB + p * 7 + o] = a[o];
    out[OFF_DOM + p] = (float)ids[best];
  }
  // recon: coalesced across lanes
  for (int t = lane; t < NBANDS; t += 32) {
    float acc = 0.f;
#pragma unroll
    for (int j = 0; j < KTOP; ++j) acc += a[j] * protos[ids[j] * NBANDS + t];
    out[OFF_REC + p * NBANDS + t] = acc;
  }
}

// ------------------------------------------------------------------
extern "C" void kernel_launch(void* const* d_in, const int* in_sizes, int n_in,
                              void* d_out, int out_size, void* d_ws, size_t ws_size,
                              hipStream_t stream) {
  const float* features = (const float*)d_in[0];
  const float* w1_s = (const float*)d_in[1];
  const float* b1_s = (const float*)d_in[2];
  const float* w2_s = (const float*)d_in[3];
  const float* b2_s = (const float*)d_in[4];
  const float* w1_a = (const float*)d_in[5];
  const float* b1_a = (const float*)d_in[6];
  const float* w2_a = (const float*)d_in[7];
  const float* b2_a = (const float*)d_in[8];
  const float* lab  = (const float*)d_in[9];
  const float* protos = (const float*)d_in[10];
  float* out = (float*)d_out;

  char* wsp = (char*)d_ws;
  auto alloc = [&](size_t bytes) {
    char* p = wsp;
    wsp += (bytes + 255) & ~(size_t)255;
    return p;
  };
  h16* lib16  = (h16*)alloc((size_t)NLIB * EMBED * 2);
  h16* fn16   = (h16*)alloc((size_t)P_PIX * EMBED * 2);
  h16* fused  = (h16*)alloc((size_t)P_PIX * FUSEDK * 2);
  h16* h1     = (h16*)alloc((size_t)P_PIX * HIDDEN * 2);
  h16* w1sT   = (h16*)alloc((size_t)HIDDEN * EMBED * 2);
  h16* w1aT   = (h16*)alloc((size_t)HIDDEN * FUSEDK * 2);
  int* cids   = (int*)alloc((size_t)P_PIX * KTOP * 4);

  // weight transposes -> f16 [N][K]
  wtrans_kernel<<<(HIDDEN * EMBED + 255) / 256, 256, 0, stream>>>(
      w1_s, w1sT, EMBED, EMBED, HIDDEN);
  wtrans_kernel<<<(HIDDEN * FUSEDK + 255) / 256, 256, 0, stream>>>(
      w1_a, w1aT, EMBED + KTOP, FUSEDK, HIDDEN);

  // normalize + convert
  prep_lib_kernel<<<NLIB / 8, 256, 0, stream>>>(lab, lib16);
  prep_feat_kernel<<<P_PIX / 8, 256, 0, stream>>>(features, fn16, fused);

  // fused cosine-sim GEMM + top-5 (the 275 GFLOP part)
  sims_topk_kernel<<<P_PIX / 64, 256, 0, stream>>>(
      fn16, lib16, out + OFF_SIMS, cids);

  // score MLP hidden (K = 256 over fused[:,0:256])
  mlp_hidden_kernel<<<dim3(P_PIX / 64, HIDDEN / 32), 256, 0, stream>>>(
      fused, FUSEDK, w1sT, EMBED, b1_s, h1);
  score_out_kernel<<<(P_PIX * KTOP + 255) / 256, 256, 0, stream>>>(
      h1, w2_s, b2_s, fused);

  // abundance MLP hidden (K = 288 over full fused)
  mlp_hidden_kernel<<<dim3(P_PIX / 64, HIDDEN / 32), 256, 0, stream>>>(
      fused, FUSEDK, w1aT, FUSEDK, b1_a, h1);

  // logits -> softmax -> abund/dominant/recon
  final_kernel<<<P_PIX / 8, 256, 0, stream>>>(
      h1, w2_a, b2_a, cids, protos, out);
}
